// transformer_layer2_19318762897743
// MI455X (gfx1250) — matrix operand order repaired and checked by reading, not compiled, not run on hardware
//
#include <hip/hip_runtime.h>
#include <math.h>

// CDNA5 / gfx1250 wave32 WMMA implementation.
typedef __attribute__((ext_vector_type(16))) _Float16 v16h;
typedef __attribute__((ext_vector_type(8)))  float    v8f;

#define NPTS  8192
#define BATCH 4
#define DIM   64
#define KNN   36

__device__ inline v8f wmma32f16(v16h a, v16h b, v8f c) {
  // 8 args: (neg_a, A, neg_b, B, c_mod, C, reuse_a, reuse_b)
  return __builtin_amdgcn_wmma_f32_16x16x32_f16(false, a, false, b, (short)0, c,
                                                false, false);
}

// A fragment (16x32 f16) from f32 source S (leading dim ld), rows mBase..+15,
// cols kBase..+31.  Rows >= mLimit are zeroed (padding).  Optional per-column
// subtraction vector (used for g - f).
// Layout (ISA 7.12.2): lane<16 -> M=lane, K = {2j,2j+1} for j<4 then K=16..23;
// lane>=16 -> same M set, K = {8..15, 24..31}.
__device__ inline v16h a_frag_f32(const float* S, int ld, int mBase, int kBase,
                                  int lane, int mLimit, const float* sub) {
  v16h a;
  const int hi = lane >> 4;
  const int m  = mBase + (lane & 15);
  const bool valid = (m < mLimit);
  #pragma unroll
  for (int j = 0; j < 8; ++j) {
    int k = kBase + 2 * j + (hi ? 8 : 0) + ((j >= 4) ? 8 : 0);
    float x0 = 0.0f, x1 = 0.0f;
    if (valid) {
      x0 = S[m * ld + k];
      x1 = S[m * ld + k + 1];
      if (sub) { x0 -= sub[k]; x1 -= sub[k + 1]; }
    }
    a[2 * j]     = (_Float16)x0;
    a[2 * j + 1] = (_Float16)x1;
  }
  return a;
}

// A fragment from an f16 source (16 rows, leading dim ld).
__device__ inline v16h a_frag_f16(const _Float16* S, int ld, int kBase, int lane) {
  v16h a;
  const int hi = lane >> 4;
  const int m  = lane & 15;
  #pragma unroll
  for (int j = 0; j < 8; ++j) {
    int k = kBase + 2 * j + (hi ? 8 : 0) + ((j >= 4) ? 8 : 0);
    a[2 * j]     = S[m * ld + k];
    a[2 * j + 1] = S[m * ld + k + 1];
  }
  return a;
}

// B fragment (32x16 KxN f16) from f32 source, rows kBase..+31, cols nBase..+15.
// Layout: lanes 0-15 hold K={0..7,16..23} (N=lane), lanes 16-31 hold K={8..15,24..31}.
__device__ inline v16h b_frag_f32(const float* S, int ld, int kBase, int nBase,
                                  int lane) {
  v16h b;
  const int hi = lane >> 4;
  const int n  = nBase + (lane & 15);
  #pragma unroll
  for (int j = 0; j < 8; ++j) {
    int k = kBase + 2 * j + (hi ? 8 : 0) + ((j >= 4) ? 8 : 0);
    b[2 * j]     = (_Float16)S[k * ld + n];
    b[2 * j + 1] = (_Float16)S[(k + 1) * ld + n];
  }
  return b;
}

// ---------------------------------------------------------------------------
// Kernel 1: brute-force KNN, top-36 by threshold-replace, sorted output.
// One thread per query point; candidate stream address is uniform across the
// block so it stays hot in L0.  Per-thread heap lives in LDS with [slot][tid]
// layout (consecutive tids -> consecutive banks, conflict free).
// ---------------------------------------------------------------------------
#define KNN_BLOCK 128
__global__ __launch_bounds__(KNN_BLOCK) void knn_kernel(
    const float* __restrict__ XYZ, int* __restrict__ IDX) {
  __shared__ float ds[KNN][KNN_BLOCK];
  __shared__ int   is_[KNN][KNN_BLOCK];
  const int b   = blockIdx.y;
  const int tid = threadIdx.x;
  const int i   = blockIdx.x * KNN_BLOCK + tid;
  const float* P = XYZ + (size_t)b * NPTS * 3;

  const float qx = P[i * 3 + 0];
  const float qy = P[i * 3 + 1];
  const float qz = P[i * 3 + 2];

  #pragma unroll
  for (int s = 0; s < KNN; ++s) { ds[s][tid] = 3.4e38f; is_[s][tid] = 0; }
  float wmax = 3.4e38f;
  int   wslot = 0;

  for (int j = 0; j < NPTS; ++j) {
    float dx = P[j * 3 + 0] - qx;
    float dy = P[j * 3 + 1] - qy;
    float dz = P[j * 3 + 2] - qz;
    float d2 = dx * dx + dy * dy + dz * dz;
    if (d2 < wmax) {
      ds[wslot][tid] = d2;
      is_[wslot][tid] = j;
      float m = -1.0f; int ms = 0;
      #pragma unroll
      for (int s = 0; s < KNN; ++s) {
        float v = ds[s][tid];
        if (v > m) { m = v; ms = s; }
      }
      wmax = m; wslot = ms;
    }
  }

  // Selection sort ascending by (dist, idx) to match top_k ordering.
  int* o = IDX + ((size_t)b * NPTS + i) * KNN;
  for (int s = 0; s < KNN; ++s) {
    float bm = 3.4e38f; int bj = 0x7fffffff; int bslot = 0;
    #pragma unroll
    for (int t = 0; t < KNN; ++t) {
      float v = ds[t][tid]; int jj = is_[t][tid];
      if (v < bm || (v == bm && jj < bj)) { bm = v; bj = jj; bslot = t; }
    }
    o[s] = bj;
    ds[bslot][tid] = 3.4e38f;
  }
}

// ---------------------------------------------------------------------------
// Kernel 2: f = relu(relu(X@W1 + b1)@W2 + b2) via two WMMA GEMMs.
// 4 waves per block, each wave owns 16 rows (64 rows per block).
// ---------------------------------------------------------------------------
__global__ __launch_bounds__(128) void mlp_kernel(
    const float* __restrict__ X, const float* __restrict__ W1,
    const float* __restrict__ b1, const float* __restrict__ W2,
    const float* __restrict__ b2, float* __restrict__ F) {
  __shared__ _Float16 Hs[4][16 * DIM];
  const int tid = threadIdx.x, wid = tid >> 5, lane = tid & 31;
  const int row0 = blockIdx.x * 64 + wid * 16;
  const float* Xr = X + (size_t)row0 * DIM;
  const int hi = lane >> 4, nl = lane & 15;

  // H = relu(X @ W1 + b1), kept in LDS as f16 (transpose through LDS).
  #pragma unroll
  for (int nT = 0; nT < 4; ++nT) {
    v8f c = {};
    #pragma unroll
    for (int ks = 0; ks < DIM; ks += 32) {
      v16h a  = a_frag_f32(Xr, DIM, 0, ks, lane, 16, (const float*)nullptr);
      v16h bb = b_frag_f32(W1, DIM, ks, nT * 16, lane);
      c = wmma32f16(a, bb, c);
    }
    int n = nT * 16 + nl;
    float bias = b1[n];
    #pragma unroll
    for (int r = 0; r < 8; ++r) {
      int m = r + 8 * hi;
      Hs[wid][m * DIM + n] = (_Float16)fmaxf(c[r] + bias, 0.0f);
    }
  }
  __syncthreads();

  // F = relu(H @ W2 + b2)
  #pragma unroll
  for (int nT = 0; nT < 4; ++nT) {
    v8f c = {};
    #pragma unroll
    for (int ks = 0; ks < DIM; ks += 32) {
      v16h a  = a_frag_f16(Hs[wid], DIM, ks, lane);
      v16h bb = b_frag_f32(W2, DIM, ks, nT * 16, lane);
      c = wmma32f16(a, bb, c);
    }
    int n = nT * 16 + nl;
    float bias = b2[n];
    #pragma unroll
    for (int r = 0; r < 8; ++r) {
      int m = r + 8 * hi;
      F[(size_t)(row0 + m) * DIM + n] = fmaxf(c[r] + bias, 0.0f);
    }
  }
}

// ---------------------------------------------------------------------------
// Kernel 3: per-(b,n) attention.  Gather G (36x64), W = (G - f)@Wr + br via
// WMMA (M padded 36->48: 3x4 tiles split over 4 waves), then the reference's
// flat-reshape softmax (windows of 36 over the flat 2304 vector, shared flat
// index for W and G), then out = att @ Ws + bs.
// ---------------------------------------------------------------------------
__global__ __launch_bounds__(128) void attn_kernel(
    const float* __restrict__ F, const int* __restrict__ IDX,
    const float* __restrict__ Wr, const float* __restrict__ br,
    const float* __restrict__ Ws, const float* __restrict__ bs,
    float* __restrict__ OUT, int writeN) {
  __shared__ float fv[DIM];
  __shared__ float G[KNN * DIM];
  __shared__ float Wm[KNN * DIM];
  __shared__ float att[DIM];

  const int b = blockIdx.y, n = blockIdx.x;
  const int tid = threadIdx.x;
  const size_t row = (size_t)b * NPTS + n;

  if (tid == 0) { __builtin_prefetch(Wr, 0, 3); __builtin_prefetch(Ws, 0, 3); }
  if (tid < DIM) fv[tid] = F[row * DIM + tid];

  // Gather neighbor features.
  for (int t = tid; t < KNN * DIM; t += 128) {
    int k = t >> 6, d = t & 63;
    int j = IDX[row * KNN + k];
    G[t] = F[((size_t)b * NPTS + j) * DIM + d];
  }
  __syncthreads();

  const int wid = tid >> 5, lane = tid & 31;
  const int hi = lane >> 4, nl = lane & 15;

  // W = (G - f) @ Wr + br   (12 MN tiles over 4 waves, K = 2 steps of 32)
  for (int t = wid; t < 12; t += 4) {
    int mT = t >> 2, nT = t & 3;
    v8f c = {};
    #pragma unroll
    for (int ks = 0; ks < DIM; ks += 32) {
      v16h a  = a_frag_f32(G, DIM, mT * 16, ks, lane, KNN, fv);
      v16h bb = b_frag_f32(Wr, DIM, ks, nT * 16, lane);
      c = wmma32f16(a, bb, c);
    }
    int nn = nT * 16 + nl;
    float bias = br[nn];
    #pragma unroll
    for (int r = 0; r < 8; ++r) {
      int m = mT * 16 + r + 8 * hi;
      if (m < KNN) Wm[m * DIM + nn] = c[r] + bias;
    }
  }
  __syncthreads();

  // Flat-reshape softmax: window d covers flat indices [d*36, d*36+36).
  if (tid < DIM) {
    const int base = tid * KNN;
    float mx = -3.4e38f;
    #pragma unroll
    for (int j = 0; j < KNN; ++j) mx = fmaxf(mx, Wm[base + j] * 0.125f);
    float s = 0.0f, acc = 0.0f;
    #pragma unroll
    for (int j = 0; j < KNN; ++j) {
      float e = __expf(Wm[base + j] * 0.125f - mx);
      s   += e;
      acc += e * G[base + j];
    }
    att[tid] = acc / s;
  }
  __syncthreads();

  // out = att @ Ws + bs
  if (tid < DIM) {
    float o = bs[tid];
    #pragma unroll 8
    for (int d = 0; d < DIM; ++d) o += att[d] * Ws[d * DIM + tid];
    OUT[row * DIM + tid] = o;
  }
  if (writeN && b == 0 && n == 0 && tid == 0)
    OUT[(size_t)BATCH * NPTS * DIM] = (float)NPTS;   // second tuple element: n
}

// ---------------------------------------------------------------------------
extern "C" void kernel_launch(void* const* d_in, const int* in_sizes, int n_in,
                              void* d_out, int out_size, void* d_ws, size_t ws_size,
                              hipStream_t stream) {
  const float* feature = (const float*)d_in[0];
  const float* xyz     = (const float*)d_in[1];
  // d_in[2] = knn_num (fixed 36, baked into the kernels)
  const float* W1 = (const float*)d_in[3];
  const float* b1 = (const float*)d_in[4];
  const float* W2 = (const float*)d_in[5];
  const float* b2 = (const float*)d_in[6];
  const float* Wr = (const float*)d_in[7];
  const float* br = (const float*)d_in[8];
  const float* Ws = (const float*)d_in[9];
  const float* bs = (const float*)d_in[10];
  float* out = (float*)d_out;

  float* Fbuf = (float*)d_ws;                                    // B*N*64 f32
  int*   IDX  = (int*)((char*)d_ws +
                       (size_t)BATCH * NPTS * DIM * sizeof(float)); // B*N*36 i32

  knn_kernel<<<dim3(NPTS / KNN_BLOCK, BATCH), KNN_BLOCK, 0, stream>>>(xyz, IDX);
  mlp_kernel<<<dim3((BATCH * NPTS) / 64), 128, 0, stream>>>(feature, W1, b1, W2,
                                                            b2, Fbuf);
  int writeN = (out_size > BATCH * NPTS * DIM) ? 1 : 0;
  attn_kernel<<<dim3(NPTS, BATCH), 128, 0, stream>>>(Fbuf, IDX, Wr, br, Ws, bs,
                                                     out, writeN);
}
